// AdaptiveProjectionHead_75874892251232
// MI455X (gfx1250) — matrix operand order repaired and checked by reading, not compiled, not run on hardware
//
#include <hip/hip_runtime.h>

#define D_DIM 2048
#define O_DIM 256
#define E_NUM 8
#define B_NUM 8192
#define DG_DIM 1024
#define BN_EPS 1e-5f

typedef __attribute__((ext_vector_type(16))) __bf16 v16bf;
typedef __attribute__((ext_vector_type(8)))  float  v8f;
typedef __attribute__((ext_vector_type(4)))  unsigned int u32x4;

union FragU { v16bf v; u32x4 u[2]; };

// A fragment (16x32 bf16, row-major source, ld = K):
// lanes 0-15 : row = l16, K = {k0..k0+7} and {k0+16..k0+23}
// lanes 16-31: row = l16, K = {k0+8..k0+15} and {k0+24..k0+31}
// caller pre-offsets base by row*ld + half*8
__device__ __forceinline__ v16bf load_a_frag(const __bf16* __restrict__ p) {
  FragU f;
  f.u[0] = *reinterpret_cast<const u32x4*>(p);
  f.u[1] = *reinterpret_cast<const u32x4*>(p + 16);
  return f.v;
}

// B fragment (32x16 bf16, from N-major (transposed) weights, ld = K):
// lanes 0-15 : col = l16, K = {k0..k0+7} and {k0+16..k0+23}
// lanes 16-31: col = l16, K = {k0+8..k0+15} and {k0+24..k0+31}
// caller pre-offsets base by col*ld + half*8
__device__ __forceinline__ v16bf load_b_frag(const __bf16* __restrict__ p) {
  FragU f;
  f.u[0] = *reinterpret_cast<const u32x4*>(p);
  f.u[1] = *reinterpret_cast<const u32x4*>(p + 16);
  return f.v;
}

// ---------------- conversion kernels ----------------
__global__ void k_cvt_bf16(const float* __restrict__ in, __bf16* __restrict__ out, size_t n) {
  size_t i = (size_t)blockIdx.x * blockDim.x + threadIdx.x;
  if (i < n) out[i] = (__bf16)in[i];
}

// out[b][c][r] = in[b][r][c]  (convert fp32 -> bf16 with transpose)
__global__ void k_cvt_transpose(const float* __restrict__ in, __bf16* __restrict__ out,
                                int rows, int cols, size_t n) {
  size_t i = (size_t)blockIdx.x * blockDim.x + threadIdx.x;
  if (i >= n) return;
  size_t rc = (size_t)rows * cols;
  size_t b = i / rc;
  size_t rem = i - b * rc;
  int c = (int)(rem / rows);
  int r = (int)(rem - (size_t)c * rows);
  out[i] = (__bf16)in[b * rc + (size_t)r * cols + c];
}

// ---------------- WMMA GEMM ----------------
// C[M,N] = A[M,K] * Bt[N,K]^T, bf16 inputs, fp32 accumulation.
// mode 0: store bf16 C                       (expert GEMM1 -> h)
// mode 1: store bf16 relu(C + bias[n])       (gate hidden)
// mode 2: atomicAdd f32 (C + bias[n]) * gateW[m*E+expert]  (expert GEMM2 + combine)
__global__ __launch_bounds__(256) void k_wmma_gemm(
    const __bf16* __restrict__ A, const __bf16* __restrict__ Bt,
    void* __restrict__ Cout, int M, int N, int K,
    int mode, const float* __restrict__ bias,
    const float* __restrict__ gateW, int expert)
{
  const int lane = threadIdx.x & 31;
  const int wave = threadIdx.x >> 5;
  const int wm = wave & 3;        // 4 waves along M
  const int wn = wave >> 2;       // 2 waves along N
  const int m0 = blockIdx.x * 128 + wm * 32;
  const int n0 = blockIdx.y * 128 + wn * 64;
  if (m0 >= M || n0 >= N) return;

  const int half = lane >> 4;
  const int l16  = lane & 15;

  const __bf16* aB0 = A  + (size_t)(m0 + l16) * K + half * 8;
  const __bf16* aB1 = aB0 + (size_t)16 * K;
  const __bf16* bB  = Bt + (size_t)(n0 + l16) * K + half * 8;
  const size_t  bS  = (size_t)16 * K;

  v8f acc00 = {}, acc01 = {}, acc02 = {}, acc03 = {};
  v8f acc10 = {}, acc11 = {}, acc12 = {}, acc13 = {};

  for (int k = 0; k < K; k += 32) {
    v16bf a0 = load_a_frag(aB0 + k);
    v16bf a1 = load_a_frag(aB1 + k);
    v16bf b0 = load_b_frag(bB + k);
    v16bf b1 = load_b_frag(bB + bS + k);
    v16bf b2 = load_b_frag(bB + 2 * bS + k);
    v16bf b3 = load_b_frag(bB + 3 * bS + k);
    acc00 = __builtin_amdgcn_wmma_f32_16x16x32_bf16(false, a0, false, b0, (short)0, acc00, false, false);
    acc01 = __builtin_amdgcn_wmma_f32_16x16x32_bf16(false, a0, false, b1, (short)0, acc01, false, false);
    acc02 = __builtin_amdgcn_wmma_f32_16x16x32_bf16(false, a0, false, b2, (short)0, acc02, false, false);
    acc03 = __builtin_amdgcn_wmma_f32_16x16x32_bf16(false, a0, false, b3, (short)0, acc03, false, false);
    acc10 = __builtin_amdgcn_wmma_f32_16x16x32_bf16(false, a1, false, b0, (short)0, acc10, false, false);
    acc11 = __builtin_amdgcn_wmma_f32_16x16x32_bf16(false, a1, false, b1, (short)0, acc11, false, false);
    acc12 = __builtin_amdgcn_wmma_f32_16x16x32_bf16(false, a1, false, b2, (short)0, acc12, false, false);
    acc13 = __builtin_amdgcn_wmma_f32_16x16x32_bf16(false, a1, false, b3, (short)0, acc13, false, false);
  }

  // C/D layout: VGPR r -> (M = r + 8*half, N = l16) within a 16x16 tile
  auto store_tile = [&](v8f& acc, int mt, int nt) {
    int ncol = n0 + nt * 16 + l16;
    #pragma unroll
    for (int r = 0; r < 8; ++r) {
      int mrow = m0 + mt * 16 + half * 8 + r;
      float v = acc[r];
      if (mode == 0) {
        reinterpret_cast<__bf16*>(Cout)[(size_t)mrow * N + ncol] = (__bf16)v;
      } else if (mode == 1) {
        v += bias[ncol];
        v = v > 0.f ? v : 0.f;
        reinterpret_cast<__bf16*>(Cout)[(size_t)mrow * N + ncol] = (__bf16)v;
      } else {
        float gw = gateW[(size_t)mrow * E_NUM + expert];
        v = (v + bias[ncol]) * gw;
        atomicAdd(reinterpret_cast<float*>(Cout) + (size_t)mrow * N + ncol, v);
      }
    }
  };
  store_tile(acc00, 0, 0); store_tile(acc01, 0, 1);
  store_tile(acc02, 0, 2); store_tile(acc03, 0, 3);
  store_tile(acc10, 1, 0); store_tile(acc11, 1, 1);
  store_tile(acc12, 1, 2); store_tile(acc13, 1, 3);
}

// ---------------- BatchNorm stats ----------------
// partial sums over a 1/32 slab of the batch per (e,d) channel
__global__ void k_bn_partial(const __bf16* __restrict__ h,
                             float* __restrict__ sumB, float* __restrict__ sqB) {
  int ch = blockIdx.x * blockDim.x + threadIdx.x;   // 0..E*D-1
  int e = ch >> 11;          // / D_DIM
  int d = ch & (D_DIM - 1);
  const int chunk = B_NUM / 32;
  int b0 = blockIdx.y * chunk;
  const __bf16* p = h + ((size_t)e * B_NUM + b0) * D_DIM + d;
  float s = 0.f, q = 0.f;
  for (int b = 0; b < chunk; ++b) {
    float v = (float)p[(size_t)b * D_DIM];
    s += v; q += v * v;
  }
  atomicAdd(&sumB[ch], s);
  atomicAdd(&sqB[ch], q);
}

__global__ void k_bn_finalize(const float* __restrict__ sumB, const float* __restrict__ sqB,
                              const float* __restrict__ gamma, const float* __restrict__ beta,
                              float* __restrict__ scaleB, float* __restrict__ shiftB) {
  int ch = blockIdx.x * blockDim.x + threadIdx.x;
  float mean = sumB[ch] * (1.f / B_NUM);
  float var  = sqB[ch] * (1.f / B_NUM) - mean * mean;
  float sc = gamma[ch] * rsqrtf(var + BN_EPS);
  scaleB[ch] = sc;
  shiftB[ch] = beta[ch] - mean * sc;
}

// in-place normalize + ReLU on bf16 h
__global__ void k_bn_apply(__bf16* __restrict__ h, const float* __restrict__ scaleB,
                           const float* __restrict__ shiftB, size_t n) {
  size_t i = (size_t)blockIdx.x * blockDim.x + threadIdx.x;
  if (i >= n) return;
  size_t e = i / ((size_t)B_NUM * D_DIM);
  int d = (int)(i & (D_DIM - 1));
  int ch = (int)e * D_DIM + d;
  float v = (float)h[i];
  v = v * scaleB[ch] + shiftB[ch];
  v = v > 0.f ? v : 0.f;
  h[i] = (__bf16)v;
}

// ---------------- gate second layer + softmax ----------------
__global__ void k_gate_softmax(const __bf16* __restrict__ g, const float* __restrict__ Wg2,
                               const float* __restrict__ bg2, float* __restrict__ gate) {
  int b = blockIdx.x * blockDim.x + threadIdx.x;
  if (b >= B_NUM) return;
  float acc[E_NUM];
  #pragma unroll
  for (int e = 0; e < E_NUM; ++e) acc[e] = bg2[e];
  const __bf16* grow = g + (size_t)b * DG_DIM;
  for (int k = 0; k < DG_DIM; ++k) {
    float gv = (float)grow[k];
    const float* wrow = Wg2 + (size_t)k * E_NUM;
    #pragma unroll
    for (int e = 0; e < E_NUM; ++e) acc[e] += gv * wrow[e];
  }
  float mx = acc[0];
  #pragma unroll
  for (int e = 1; e < E_NUM; ++e) mx = acc[e] > mx ? acc[e] : mx;
  float s = 0.f;
  #pragma unroll
  for (int e = 0; e < E_NUM; ++e) { acc[e] = expf(acc[e] - mx); s += acc[e]; }
  float inv = 1.f / s;
  #pragma unroll
  for (int e = 0; e < E_NUM; ++e) gate[(size_t)b * E_NUM + e] = acc[e] * inv;
}

// ---------------- host launcher ----------------
extern "C" void kernel_launch(void* const* d_in, const int* in_sizes, int n_in,
                              void* d_out, int out_size, void* d_ws, size_t ws_size,
                              hipStream_t stream) {
  (void)in_sizes; (void)n_in; (void)out_size; (void)ws_size;
  const float* x     = (const float*)d_in[0];
  const float* W1    = (const float*)d_in[1];
  // d_in[2] = b1: cancels exactly under training-mode BatchNorm (constant per channel)
  const float* gamma = (const float*)d_in[3];
  const float* beta  = (const float*)d_in[4];
  const float* W2    = (const float*)d_in[5];
  const float* b2    = (const float*)d_in[6];
  const float* Wg1   = (const float*)d_in[7];
  const float* bg1   = (const float*)d_in[8];
  const float* Wg2   = (const float*)d_in[9];
  const float* bg2   = (const float*)d_in[10];
  float* out = (float*)d_out;

  char* w = (char*)d_ws;
  size_t off = 0;
  auto alloc = [&](size_t bytes) -> void* {
    void* p = w + off;
    off = (off + bytes + 255) & ~(size_t)255;
    return p;
  };
  __bf16* x_bf  = (__bf16*)alloc((size_t)B_NUM * D_DIM * 2);
  __bf16* W1t   = (__bf16*)alloc((size_t)E_NUM * D_DIM * D_DIM * 2);
  __bf16* W2t   = (__bf16*)alloc((size_t)E_NUM * O_DIM * D_DIM * 2);
  __bf16* Wg1t  = (__bf16*)alloc((size_t)DG_DIM * D_DIM * 2);
  __bf16* gbuf  = (__bf16*)alloc((size_t)B_NUM * DG_DIM * 2);
  float*  gate  = (float*)alloc((size_t)B_NUM * E_NUM * 4);
  __bf16* hbuf  = (__bf16*)alloc((size_t)E_NUM * B_NUM * D_DIM * 2);
  float*  sumB  = (float*)alloc((size_t)E_NUM * D_DIM * 4);
  float*  sqB   = (float*)alloc((size_t)E_NUM * D_DIM * 4);
  float*  scaleB= (float*)alloc((size_t)E_NUM * D_DIM * 4);
  float*  shiftB= (float*)alloc((size_t)E_NUM * D_DIM * 4);

  // precision staging: fp32 -> bf16, weights pre-transposed to [N][K]
  { size_t n = (size_t)B_NUM * D_DIM;
    k_cvt_bf16<<<(unsigned)((n + 255) / 256), 256, 0, stream>>>(x, x_bf, n); }
  { size_t n = (size_t)E_NUM * D_DIM * D_DIM;
    k_cvt_transpose<<<(unsigned)((n + 255) / 256), 256, 0, stream>>>(W1, W1t, D_DIM, D_DIM, n); }
  { size_t n = (size_t)E_NUM * D_DIM * O_DIM;
    k_cvt_transpose<<<(unsigned)((n + 255) / 256), 256, 0, stream>>>(W2, W2t, D_DIM, O_DIM, n); }
  { size_t n = (size_t)D_DIM * DG_DIM;
    k_cvt_transpose<<<(unsigned)((n + 255) / 256), 256, 0, stream>>>(Wg1, Wg1t, D_DIM, DG_DIM, n); }

  // gate: WMMA GEMM with fused bias+ReLU, then small per-row softmax kernel
  k_wmma_gemm<<<dim3(B_NUM / 128, DG_DIM / 128), 256, 0, stream>>>(
      x_bf, Wg1t, gbuf, B_NUM, DG_DIM, D_DIM, 1, bg1, nullptr, 0);
  k_gate_softmax<<<B_NUM / 256, 256, 0, stream>>>(gbuf, Wg2, bg2, gate);

  // expert GEMM1: h[e] = x @ W1[e]  (b1 skipped: cancels in BN)
  for (int e = 0; e < E_NUM; ++e)
    k_wmma_gemm<<<dim3(B_NUM / 128, D_DIM / 128), 256, 0, stream>>>(
        x_bf, W1t + (size_t)e * D_DIM * D_DIM,
        hbuf + (size_t)e * B_NUM * D_DIM,
        B_NUM, D_DIM, D_DIM, 0, nullptr, nullptr, 0);

  // BatchNorm (training-mode batch stats) folded to scale/shift, fused ReLU in-place
  (void)hipMemsetAsync(sumB, 0, (size_t)E_NUM * D_DIM * 4, stream);
  (void)hipMemsetAsync(sqB,  0, (size_t)E_NUM * D_DIM * 4, stream);
  k_bn_partial<<<dim3(E_NUM * D_DIM / 256, 32), 256, 0, stream>>>(hbuf, sumB, sqB);
  k_bn_finalize<<<E_NUM * D_DIM / 256, 256, 0, stream>>>(sumB, sqB, gamma, beta, scaleB, shiftB);
  { size_t n = (size_t)E_NUM * B_NUM * D_DIM;
    k_bn_apply<<<(unsigned)((n + 255) / 256), 256, 0, stream>>>(hbuf, scaleB, shiftB, n); }

  // expert GEMM2 fused with +b2, gate weighting, and combine (atomic f32 adds)
  (void)hipMemsetAsync(out, 0, (size_t)B_NUM * O_DIM * 4, stream);
  for (int e = 0; e < E_NUM; ++e)
    k_wmma_gemm<<<dim3(B_NUM / 128, O_DIM / 128), 256, 0, stream>>>(
        hbuf + (size_t)e * B_NUM * D_DIM,
        W2t + (size_t)e * O_DIM * D_DIM,
        out, B_NUM, O_DIM, D_DIM, 2, b2 + (size_t)e * O_DIM, gate, e);
}
